// MultiHeadAttention_6622839571233
// MI455X (gfx1250) — matrix operand order repaired and checked by reading, not compiled, not run on hardware
//
#include <hip/hip_runtime.h>
#include <stddef.h>
#include <stdint.h>

// ---------------- problem constants ----------------
#define Bsz 2
#define Sq  2048
#define Dm  1024
#define Hn  16
#define DKc 64
#define Mrows (Bsz*Sq)   // 4096

// ---------------- types ----------------
typedef __bf16 bf16_t;
typedef __attribute__((ext_vector_type(16))) __bf16 v16bf;
typedef __attribute__((ext_vector_type(8)))  __bf16 v8bf;
typedef __attribute__((ext_vector_type(4)))  __bf16 v4bf;
typedef __attribute__((ext_vector_type(8)))  float  v8f;
typedef __attribute__((ext_vector_type(4)))  float  v4f;
typedef __attribute__((ext_vector_type(4)))  unsigned int v4u;
typedef __attribute__((ext_vector_type(8)))  int    v8i;
typedef __attribute__((ext_vector_type(4)))  int    v4i;

// ---------------- WMMA helper ----------------
__device__ __forceinline__ v8f wmma_bf16(v16bf a, v16bf b, v8f c) {
  return __builtin_amdgcn_wmma_f32_16x16x32_bf16(
      false, a, false, b, (short)0, c, false, false);
}

// A-matrix 16x32 bf16 fragment, contiguous k order (B frags use the same):
//   lane L: m = L&15, kh = L>>4
//   elements 0..15 = K = kh*16 + 0..15
__device__ __forceinline__ v16bf load_afrag_bf16(const bf16_t* row, int kh) {
  v8bf c0 = *(const v8bf*)(row + kh * 16);
  v8bf c1 = *(const v8bf*)(row + kh * 16 + 8);
  v16bf r;
#pragma unroll
  for (int t = 0; t < 8; ++t) { r[t] = c0[t]; r[8 + t] = c1[t]; }
  return r;
}

// ---------------- TDM: 2D bf16 tile -> LDS (D# per ISA 8.3/8.4) ----------------
// tile_k elems (dim0, contiguous), tile_rows (dim1), row_stride in elements.
// Toolchain here is the clang-23 6-arg builtin:
//   (uint32x4 g0, int32x8 g1, int32x4 g2, int32x4 g3, int32x8 pad, i32 cpol)
__device__ __forceinline__ void tdm_load_2d_bf16(
    unsigned lds_off, const bf16_t* gptr,
    unsigned tile_k, unsigned tile_rows,
    unsigned tensor_k, unsigned tensor_rows, unsigned row_stride)
{
  unsigned long long ga = (unsigned long long)(uintptr_t)gptr;
  v4u g0;
  g0[0] = 1u;                                   // count=1, user desc, no gather
  g0[1] = lds_off;                              // lds_addr [63:32]
  g0[2] = (unsigned)(ga & 0xFFFFFFFFu);         // global_addr [95:64]
  g0[3] = (unsigned)((ga >> 32) & 0x1FFFFFFu)   // global_addr hi
        | (2u << 30);                           // type=2 ("image")
  v8i g1;
  g1[0] = (int)(1u << 16);                      // data_size=1 (2 bytes)
  g1[1] = (int)(tensor_k << 16);                // tensor_dim0 lo16 @ bits 63:48
  g1[2] = (int)((tensor_k >> 16) | (tensor_rows << 16));  // dim0 hi | dim1 lo
  g1[3] = (int)((tensor_rows >> 16) | (tile_k << 16));    // dim1 hi | tile_dim0
  g1[4] = (int)tile_rows;                       // tile_dim1 (tile_dim2 = 0)
  g1[5] = (int)row_stride;                      // tensor_dim0_stride lo32
  g1[6] = 0;                                    // stride0 hi | stride1 lo
  g1[7] = 0;
  v4i z4 = {0, 0, 0, 0};
  v8i z8 = {0, 0, 0, 0, 0, 0, 0, 0};
  __builtin_amdgcn_tensor_load_to_lds(g0, g1, z4, z4, z8, 0);
}

// =====================================================================
// Kernel 0: one-time fp32 -> bf16 conversion of the 4 weight matrices.
// Wbf layout: [4][D*D] bf16 (Wq, Wk, Wv, Wo), row-major preserved.
// =====================================================================
__global__ __launch_bounds__(256) void cvt_weights_kernel(
    const float* __restrict__ Wq, const float* __restrict__ Wk,
    const float* __restrict__ Wv, const float* __restrict__ Wo,
    bf16_t* __restrict__ Wbf)
{
  const int z = blockIdx.y;
  const float* src = (z == 0) ? Wq : (z == 1) ? Wk : (z == 2) ? Wv : Wo;
  bf16_t* dst = Wbf + (size_t)z * Dm * Dm;
  const size_t i = ((size_t)blockIdx.x * 256 + threadIdx.x) * 4;
  v4f f = *(const v4f*)(src + i);
  v4bf o;
#pragma unroll
  for (int t = 0; t < 4; ++t) o[t] = (bf16_t)f[t];
  *(v4bf*)(dst + i) = o;
}

// =====================================================================
// Kernel 1: fused QKV projection.  Y = X @ W^T  (NT GEMM), X fp32, W bf16.
//   z = 0: Q -> qp  [B,H,S,DK] bf16 (scaled 1/sqrt(DK))
//   z = 1: K -> kp  [B,H,S,DK] bf16
//   z = 2: V -> vpT [B,H,DK,S] bf16 (transposed for contiguous PV B-frags)
// W tile staged bf16 via TDM, double-buffered against the WMMA chain.
// =====================================================================
__global__ __launch_bounds__(256) void proj_qkv_kernel(
    const float* __restrict__ Q, const float* __restrict__ K,
    const float* __restrict__ V, const bf16_t* __restrict__ Wbf,
    bf16_t* __restrict__ qp, bf16_t* __restrict__ kp,
    bf16_t* __restrict__ vpT)
{
  __shared__ __attribute__((aligned(64))) bf16_t wtile[2][128 * 32]; // 16 KB

  const int z = blockIdx.z;
  const float*  X = (z == 0) ? Q : (z == 1) ? K : V;
  const bf16_t* W = Wbf + (size_t)z * Dm * Dm;

  const int tileM = blockIdx.x * 128;
  const int tileN = blockIdx.y * 128;
  const int tid  = threadIdx.x;
  const int lane = tid & 31;
  const int wave = tid >> 5;
  const int lm   = lane & 15;
  const int kh   = lane >> 4;
  const int m0   = tileM + wave * 16;

  // LDS offsets (generic shared pointer: low 32 bits = LDS offset)
  const unsigned lds0 = (unsigned)(uintptr_t)(void*)&wtile[0][0];
  const unsigned lds1 = (unsigned)(uintptr_t)(void*)&wtile[1][0];

  v8f acc[8] = {};

  // prologue: DMA first W tile (rows tileN..+127, k 0..31)
  if (tid == 0) {
    tdm_load_2d_bf16(lds0, W + (size_t)tileN * Dm, 32, 128, Dm, Dm, Dm);
    __builtin_amdgcn_s_wait_tensorcnt((short)0);
  }
  __syncthreads();

  for (int k0 = 0; k0 < Dm; k0 += 32) {
    const int cur = (k0 >> 5) & 1;
    const bf16_t* wt = &wtile[cur][0];

    // kick DMA for next k tile into the other buffer
    if (tid == 0 && (k0 + 32) < Dm) {
      tdm_load_2d_bf16(cur ? lds0 : lds1,
                       W + (size_t)tileN * Dm + (k0 + 32), 32, 128, Dm, Dm, Dm);
    }

    // ---- A fragment: 16x32 of X, fp32 -> bf16 in regs ----
    const float* arow = X + (size_t)(m0 + lm) * Dm + k0;
    __builtin_prefetch(arow + 64, 0, 1);
    v16bf a;
    {
      v4f c0 = *(const v4f*)(arow + kh * 16);
      v4f c1 = *(const v4f*)(arow + kh * 16 + 4);
      v4f c2 = *(const v4f*)(arow + kh * 16 + 8);
      v4f c3 = *(const v4f*)(arow + kh * 16 + 12);
#pragma unroll
      for (int t = 0; t < 4; ++t) {
        a[t]      = (bf16_t)c0[t];
        a[4 + t]  = (bf16_t)c1[t];
        a[8 + t]  = (bf16_t)c2[t];
        a[12 + t] = (bf16_t)c3[t];
      }
    }

    // ---- preload all 8 B fragments, then run the WMMA chain ----
    v16bf bfr[8];
#pragma unroll
    for (int j = 0; j < 8; ++j)
      bfr[j] = *(const v16bf*)&wt[(j * 16 + lm) * 32 + kh * 16];
#pragma unroll
    for (int j = 0; j < 8; ++j)
      acc[j] = wmma_bf16(a, bfr[j], acc[j]);

    if (tid == 0) __builtin_amdgcn_s_wait_tensorcnt((short)0);
    __syncthreads();
  }

  // ---- epilogue: scatter to head-major bf16 layouts ----
  const float qscale = (z == 0) ? 0.125f : 1.0f;  // 1/sqrt(64) folded into Q
#pragma unroll
  for (int j = 0; j < 8; ++j) {
    const int col = tileN + j * 16 + lm;
    const int h = col >> 6;
    const int d = col & 63;
#pragma unroll
    for (int r = 0; r < 8; ++r) {
      const int gm = m0 + r + kh * 8;
      const int b  = gm >> 11;
      const int s  = gm & (Sq - 1);
      const bf16_t val = (bf16_t)(acc[j][r] * qscale);
      if (z == 0)
        qp[(((size_t)(b * Hn + h) * Sq + s) * DKc) + d] = val;
      else if (z == 1)
        kp[(((size_t)(b * Hn + h) * Sq + s) * DKc) + d] = val;
      else
        vpT[(((size_t)(b * Hn + h) * DKc + d) * Sq) + s] = val;
    }
  }
}

// =====================================================================
// Kernel 2: causal flash attention per (b,h).
// Block = 128 thr (4 waves); wave owns 16 q rows; iterates 32-key tiles.
// Per tile: 4 WMMA (QK^T) + online softmax + 4 WMMA (PV).
// =====================================================================
__global__ __launch_bounds__(128) void flash_attn_kernel(
    const bf16_t* __restrict__ qp, const bf16_t* __restrict__ kp,
    const bf16_t* __restrict__ vpT, bf16_t* __restrict__ attn)
{
  __shared__ __attribute__((aligned(32))) bf16_t pbuf[4][16 * 32];

  const int bh   = blockIdx.y;
  const int b    = bh >> 4;
  const int h    = bh & 15;
  const int tid  = threadIdx.x;
  const int lane = tid & 31;
  const int wave = tid >> 5;
  const int lm   = lane & 15;
  const int kh   = lane >> 4;
  const int q0   = blockIdx.x * 64 + wave * 16;

  const bf16_t* qbase = qp  + (size_t)bh * Sq * DKc;
  const bf16_t* kbase = kp  + (size_t)bh * Sq * DKc;
  const bf16_t* vbase = vpT + (size_t)bh * DKc * Sq;

  const bf16_t* qrow = qbase + (size_t)(q0 + lm) * DKc;
  const v16bf aq0 = load_afrag_bf16(qrow, kh);
  const v16bf aq1 = load_afrag_bf16(qrow + 32, kh);

  float m_r[8], l_r[8];
#pragma unroll
  for (int r = 0; r < 8; ++r) { m_r[r] = -1e30f; l_r[r] = 0.0f; }
  v8f o[4] = {};

  bf16_t* pw = &pbuf[wave][0];

  for (int kc = 0; kc <= q0 + 15; kc += 32) {
    const bf16_t* kr0 = kbase + (size_t)(kc + lm) * DKc + kh * 16;
    const bf16_t* kr1 = kbase + (size_t)(kc + 16 + lm) * DKc + kh * 16;
    v8f sc0 = {}, sc1 = {};
    sc0 = wmma_bf16(aq0, *(const v16bf*)(kr0),      sc0);
    sc0 = wmma_bf16(aq1, *(const v16bf*)(kr0 + 32), sc0);
    sc1 = wmma_bf16(aq0, *(const v16bf*)(kr1),      sc1);
    sc1 = wmma_bf16(aq1, *(const v16bf*)(kr1 + 32), sc1);

    float alpha[8];
#pragma unroll
    for (int r = 0; r < 8; ++r) {
      const int qi = q0 + r + kh * 8;
      float s0 = sc0[r];
      float s1 = sc1[r];
      if (kc + lm > qi)      s0 = -1e30f;
      if (kc + 16 + lm > qi) s1 = -1e30f;

      float mt = fmaxf(s0, s1);
#pragma unroll
      for (int x = 1; x < 16; x <<= 1) mt = fmaxf(mt, __shfl_xor(mt, x, 32));
      const float mn = fmaxf(m_r[r], mt);
      const float al = __expf(m_r[r] - mn);
      const float e0 = __expf(s0 - mn);
      const float e1 = __expf(s1 - mn);
      float rs = e0 + e1;
#pragma unroll
      for (int x = 1; x < 16; x <<= 1) rs += __shfl_xor(rs, x, 32);
      l_r[r] = l_r[r] * al + rs;
      m_r[r] = mn;
      alpha[r] = al;

      bf16_t* pr = pw + (r + kh * 8) * 32;
      pr[lm]      = (bf16_t)e0;
      pr[16 + lm] = (bf16_t)e1;
    }

#pragma unroll
    for (int j = 0; j < 4; ++j)
#pragma unroll
      for (int r = 0; r < 8; ++r) o[j][r] *= alpha[r];

    const v16bf pa = load_afrag_bf16(pw + lm * 32, kh);

#pragma unroll
    for (int j = 0; j < 4; ++j) {
      const bf16_t* vr = vbase + (size_t)(j * 16 + lm) * Sq + kc + kh * 16;
      o[j] = wmma_bf16(pa, *(const v16bf*)vr, o[j]);
    }
  }

#pragma unroll
  for (int j = 0; j < 4; ++j) {
    const int col = h * DKc + j * 16 + lm;
#pragma unroll
    for (int r = 0; r < 8; ++r) {
      const int s = q0 + r + kh * 8;
      const float v = o[j][r] / l_r[r];
      attn[((size_t)(b * Sq + s)) * Dm + col] = (bf16_t)v;
    }
  }
}

// =====================================================================
// Kernel 3: output projection  out = attn(bf16) @ Wo^T -> fp32
// Same blocking as kernel 1; TDM-staged bf16 Wo, preloaded B frags.
// =====================================================================
__global__ __launch_bounds__(256) void out_proj_kernel(
    const bf16_t* __restrict__ Ain, const bf16_t* __restrict__ Wobf,
    float* __restrict__ out)
{
  __shared__ __attribute__((aligned(64))) bf16_t wtile[2][128 * 32];

  const int tileM = blockIdx.x * 128;
  const int tileN = blockIdx.y * 128;
  const int tid  = threadIdx.x;
  const int lane = tid & 31;
  const int wave = tid >> 5;
  const int lm   = lane & 15;
  const int kh   = lane >> 4;
  const int m0   = tileM + wave * 16;

  const unsigned lds0 = (unsigned)(uintptr_t)(void*)&wtile[0][0];
  const unsigned lds1 = (unsigned)(uintptr_t)(void*)&wtile[1][0];

  v8f acc[8] = {};

  if (tid == 0) {
    tdm_load_2d_bf16(lds0, Wobf + (size_t)tileN * Dm, 32, 128, Dm, Dm, Dm);
    __builtin_amdgcn_s_wait_tensorcnt((short)0);
  }
  __syncthreads();

  for (int k0 = 0; k0 < Dm; k0 += 32) {
    const int cur = (k0 >> 5) & 1;
    const bf16_t* wt = &wtile[cur][0];

    if (tid == 0 && (k0 + 32) < Dm) {
      tdm_load_2d_bf16(cur ? lds0 : lds1,
                       Wobf + (size_t)tileN * Dm + (k0 + 32), 32, 128, Dm, Dm, Dm);
    }

    const bf16_t* arow = Ain + (size_t)(m0 + lm) * Dm + k0;
    __builtin_prefetch(arow + 128, 0, 1);
    const v16bf a = load_afrag_bf16(arow, kh);

    v16bf bfr[8];
#pragma unroll
    for (int j = 0; j < 8; ++j)
      bfr[j] = *(const v16bf*)&wt[(j * 16 + lm) * 32 + kh * 16];
#pragma unroll
    for (int j = 0; j < 8; ++j)
      acc[j] = wmma_bf16(a, bfr[j], acc[j]);

    if (tid == 0) __builtin_amdgcn_s_wait_tensorcnt((short)0);
    __syncthreads();
  }

#pragma unroll
  for (int j = 0; j < 8; ++j) {
    const int col = tileN + j * 16 + lm;
#pragma unroll
    for (int r = 0; r < 8; ++r) {
      const int gm = m0 + r + kh * 8;
      out[(size_t)gm * Dm + col] = acc[j][r];
    }
  }
}

// =====================================================================
// Launch: cvt weights -> QKV proj -> flash attention -> out proj.
// ws: qp, kp (bf16 [B,H,S,DK]), vpT (bf16 [B,H,DK,S]), attn (bf16 [B,S,D]),
//     Wbf (bf16 [4][D*D])  => 40 MB total.
// =====================================================================
extern "C" void kernel_launch(void* const* d_in, const int* in_sizes, int n_in,
                              void* d_out, int out_size, void* d_ws, size_t ws_size,
                              hipStream_t stream)
{
  const float* Q  = (const float*)d_in[0];
  const float* K  = (const float*)d_in[1];
  const float* V  = (const float*)d_in[2];
  const float* Wq = (const float*)d_in[3];
  const float* Wk = (const float*)d_in[4];
  const float* Wv = (const float*)d_in[5];
  const float* Wo = (const float*)d_in[6];
  float* out = (float*)d_out;

  const size_t per = (size_t)Bsz * Hn * Sq * DKc;   // 4M elements
  bf16_t* qp   = (bf16_t*)d_ws;
  bf16_t* kp   = qp + per;
  bf16_t* vpT  = kp + per;
  bf16_t* attn = vpT + per;
  bf16_t* Wbf  = attn + per;                        // 4 x Dm*Dm bf16

  dim3 gw(Dm * Dm / (4 * 256), 4, 1);
  cvt_weights_kernel<<<gw, dim3(256), 0, stream>>>(Wq, Wk, Wv, Wo, Wbf);

  dim3 gp(Mrows / 128, Dm / 128, 3);
  proj_qkv_kernel<<<gp, dim3(256), 0, stream>>>(Q, K, V, Wbf, qp, kp, vpT);

  dim3 ga(Sq / 64, Bsz * Hn, 1);
  flash_attn_kernel<<<ga, dim3(128), 0, stream>>>(qp, kp, vpT, attn);

  dim3 go(Mrows / 128, Dm / 128, 1);
  out_proj_kernel<<<go, dim3(256), 0, stream>>>(attn, Wbf + (size_t)3 * Dm * Dm, out);
}
